// SimpleRetention_72533407695346
// MI455X (gfx1250) — matrix operand order repaired and checked by reading, not compiled, not run on hardware
//
#include <hip/hip_runtime.h>
#include <hip/hip_bf16.h>
#include <math.h>

#define HIDDEN 512
#define HEAD   64
#define SEQ    2048
#define CHUNK  256
#define BATCH  8
#define GAMMA_F     0.96875f
#define INV_GAMMA_F 1.0322580645161290f      // 1/0.96875
#define LOG2GAMMA (-0.04580368961312478f)    // log2(0.96875)

typedef __attribute__((ext_vector_type(16))) _Float16 v16h;
typedef __attribute__((ext_vector_type(8)))  float    v8f;
typedef __attribute__((ext_vector_type(2)))  float    v2f;

// ---------------------------------------------------------------------------
// Kernel 1: Q/K/V projections via V_WMMA_F32_16X16X4_F32 (full f32), + xPos.
// Grid: BATCH * (SEQ/16) blocks of 384 threads (12 wave32).
// Wave w: matrix = w/4 (0=Q,1=K,2=V), N-tile = w%4. Each wave owns one
// 16x16 C tile and contracts K=512 in 128 steps of 4.
// ---------------------------------------------------------------------------
__global__ __launch_bounds__(384) void retnet_proj_xpos(
    const float* __restrict__ X,
    const float* __restrict__ WQ,
    const float* __restrict__ WK,
    const float* __restrict__ WV,
    _Float16* __restrict__ Qh,
    _Float16* __restrict__ Kh,
    _Float16* __restrict__ Vh)
{
    constexpr int XS = HIDDEN + 4;   // padded f32 stride: spread LDS banks
    __shared__ __align__(16) float sX[16 * XS];   // 33 KB
    __shared__ float sQ[16 * HEAD];
    __shared__ float sK[16 * HEAD];

    const int b    = blockIdx.x / (SEQ / 16);
    const int rt   = blockIdx.x % (SEQ / 16);
    const int row0 = rt * 16;
    const int t    = threadIdx.x;
    const int wave = t >> 5;
    const int lane = t & 31;
    const int Lm   = lane & 15;
    const int Lh   = lane >> 4;
    const int mat  = wave >> 2;     // 0=Q, 1=K, 2=V
    const int nt   = wave & 3;      // N tile (16 cols)

    // --- stage X rows (16 x 512 f32) via float4 (B128) ---
    const float* Xb = X + ((size_t)b * SEQ + row0) * HIDDEN;
    for (int idx = t; idx < 16 * (HIDDEN / 4); idx += 384) {
        const int r = idx >> 7, c = idx & 127;
        *reinterpret_cast<float4*>(&sX[r * XS + c * 4]) =
            reinterpret_cast<const float4*>(Xb)[r * (HIDDEN / 4) + c];
    }
    __syncthreads();

    const float* W = (mat == 0) ? WQ : (mat == 1) ? WK : WV;
    const int col = nt * 16 + Lm;

    // A layout (16x4 f32): m = lane%16, k = v + 2*(lane/16)
    // B layout (4x16 f32): n = lane%16, k = v + 2*(lane/16)
    v8f c8 = (v8f){0.f,0.f,0.f,0.f,0.f,0.f,0.f,0.f};
    for (int k4 = 0; k4 < HIDDEN / 4; ++k4) {
        const int kb = k4 * 4 + 2 * Lh;
        v2f a, bw;
        a[0] = sX[Lm * XS + kb];
        a[1] = sX[Lm * XS + kb + 1];
        bw[0] = W[(size_t)kb * HEAD + col];
        bw[1] = W[(size_t)(kb + 1) * HEAD + col];
        c8 = __builtin_amdgcn_wmma_f32_16x16x4_f32(false, a, false, bw,
                                                   (short)0, c8, false, false);
    }

    // --- deposit C tile: V straight to global f16; Q/K to LDS for xPos ---
    if (mat == 2) {
        _Float16* Vb = Vh + ((size_t)b * SEQ + row0) * HEAD;
#pragma unroll
        for (int v = 0; v < 8; ++v)
            Vb[(size_t)(v + 8 * Lh) * HEAD + col] = (_Float16)c8[v];
    } else {
        float* dst = (mat == 0) ? sQ : sK;
#pragma unroll
        for (int v = 0; v < 8; ++v)
            dst[(v + 8 * Lh) * HEAD + col] = c8[v];
    }
    __syncthreads();

    // --- xPos: 16 rows x 32 even/odd pairs = 512 tasks ---
    for (int task = t; task < 16 * 32; task += 384) {
        const int row = task >> 5;
        const int m   = task & 31;
        const int l   = row0 + row;

        const float sbase = (2.0f * (float)m + 0.4f * (float)HEAD) / (1.4f * (float)HEAD);
        const float scale = powf(sbase, (float)l / 512.0f);
        const float invf  = powf(10000.0f, -((float)m) / 32.0f);
        const float ang   = (float)l * invf;
        const float sn = sinf(ang);
        const float cs = cosf(ang);

        const float sq = scale;          // Q: upscale
        const float sk = 1.0f / scale;   // K: downscale

        const float x0q = sQ[row * HEAD + 2 * m];
        const float x1q = sQ[row * HEAD + 2 * m + 1];
        const float x0k = sK[row * HEAD + 2 * m];
        const float x1k = sK[row * HEAD + 2 * m + 1];

        const size_t o = ((size_t)b * SEQ + l) * HEAD;
        Qh[o + 2 * m]     = (_Float16)((x0q * cs - x1q * sn) * sq);
        Qh[o + 2 * m + 1] = (_Float16)((x1q * cs + x0q * sn) * sq);
        Kh[o + 2 * m]     = (_Float16)((x0k * cs - x1k * sn) * sk);
        Kh[o + 2 * m + 1] = (_Float16)((x1k * cs + x0k * sn) * sk);
    }
}

// ---------------------------------------------------------------------------
// Kernel 2: fused retention  O = (Q K^T ⊙ D) V   via v_wmma_f32_16x16x32_f16.
// Grid: BATCH * (SEQ/64) blocks of 128 threads (4 wave32).
// Wave w owns query rows [qbase + 16w, +16). All waves share the same chunk
// -> same key extent kend -> uniform barriers and loop trip counts.
// ---------------------------------------------------------------------------
__global__ __launch_bounds__(128) void retnet_attn_wmma(
    const _Float16* __restrict__ Qh,
    const _Float16* __restrict__ Kh,
    const _Float16* __restrict__ Vh,
    float* __restrict__ Out)
{
    constexpr int KSTR = HEAD + 8;  // padded LDS row stride (halves); 144 B, 16B-aligned
    constexpr int SSTR = 40;        // padded stride for 16x32 S tile

    __shared__ __align__(16) _Float16 sQ[64 * KSTR];
    __shared__ __align__(16) _Float16 sK[32 * KSTR];
    __shared__ __align__(16) _Float16 sV[32 * KSTR];
    __shared__ __align__(16) _Float16 sS[4][16 * SSTR];

    const int b     = blockIdx.x >> 5;        // / (SEQ/64)
    const int qb    = blockIdx.x & 31;
    const int qbase = qb * 64;
    const int kend  = ((qbase / CHUNK) + 1) * CHUNK;  // all j < kend kept
    const int nIter = kend / 32;

    const int t    = threadIdx.x;
    const int wave = t >> 5;
    const int lane = t & 31;
    const int Lm   = lane & 15;   // N (B/C cols) / M (A rows)
    const int Lh   = lane >> 4;   // upper-half selector

    // --- stage Q block (64 x 64, f16) into LDS: B128 transactions ---
    const _Float16* Qb = Qh + ((size_t)b * SEQ + qbase) * HEAD;
    {
        const uint4* g = reinterpret_cast<const uint4*>(Qb);   // 8 uint4 per row
        for (int idx = t; idx < 64 * 8; idx += 128) {
            const int r = idx >> 3, c = idx & 7;
            *reinterpret_cast<uint4*>(&sQ[r * KSTR + c * 8]) = g[r * 8 + c];
        }
    }
    __syncthreads();

    // --- build Q A-fragments (16x32 each, head dims [0,32) and [32,64)) ---
    // A layout: m = lane%16 ; k = 16*(lane/16) + 2*v + h
    v16h aq0, aq1;
#pragma unroll
    for (int v = 0; v < 8; ++v) {
#pragma unroll
        for (int h = 0; h < 2; ++h) {
            const int kk = 16 * Lh + 2 * v + h;
            aq0[2 * v + h] = sQ[(wave * 16 + Lm) * KSTR + kk];
            aq1[2 * v + h] = sQ[(wave * 16 + Lm) * KSTR + 32 + kk];
        }
    }

    v8f o[4];
#pragma unroll
    for (int n = 0; n < 4; ++n) o[n] = (v8f){0.f,0.f,0.f,0.f,0.f,0.f,0.f,0.f};

    const int i0 = qbase + wave * 16 + 8 * Lh;   // seq row of accumulator v=0

    for (int it = 0; it < nIter; ++it) {
        const int kb = it * 32;
        __syncthreads();  // previous sK/sV/sS fully consumed

        const _Float16* Kb = Kh + ((size_t)b * SEQ + kb) * HEAD;
        const _Float16* Vb = Vh + ((size_t)b * SEQ + kb) * HEAD;
        {
            const uint4* gk = reinterpret_cast<const uint4*>(Kb);
            const uint4* gv = reinterpret_cast<const uint4*>(Vb);
            for (int idx = t; idx < 32 * 8; idx += 128) {   // 2 iters/thread
                const int r = idx >> 3, c = idx & 7;
                *reinterpret_cast<uint4*>(&sK[r * KSTR + c * 8]) = gk[r * 8 + c];
                *reinterpret_cast<uint4*>(&sV[r * KSTR + c * 8]) = gv[r * 8 + c];
            }
        }
        if (it + 1 < nIter) {  // global_prefetch_b8 of next tiles
            __builtin_prefetch(Kb + 32 * HEAD, 0, 0);
            __builtin_prefetch(Vb + 32 * HEAD, 0, 0);
        }
        __syncthreads();

        // --- S = Q K^T over 32 keys, as two 16x16 tiles; decay in f32 ---
#pragma unroll
        for (int s = 0; s < 2; ++s) {
            // B layout (32x16): n = lane%16 ; k = 16*(lane/16) + 2*v + h
            v16h bk0, bk1;
#pragma unroll
            for (int v = 0; v < 8; ++v) {
#pragma unroll
                for (int h = 0; h < 2; ++h) {
                    const int kd  = 16 * Lh + 2 * v + h;     // head dim
                    const int key = s * 16 + Lm;             // key within tile
                    bk0[2 * v + h] = sK[key * KSTR + kd];
                    bk1[2 * v + h] = sK[key * KSTR + 32 + kd];
                }
            }
            v8f sc = (v8f){0.f,0.f,0.f,0.f,0.f,0.f,0.f,0.f};
            sc = __builtin_amdgcn_wmma_f32_16x16x32_f16(false, aq0, false, bk0,
                                                        (short)0, sc, false, false);
            sc = __builtin_amdgcn_wmma_f32_16x16x32_f16(false, aq1, false, bk1,
                                                        (short)0, sc, false, false);

            // decay: gamma^|i-j| = min(gamma^(i-j), gamma^(j-i)); along v the
            // exponent steps by +1, so 2 raw v_exp_f32 + mul/min recurrences.
            // |base| < 2048 -> exp2 argument within +/-94: no over/underflow.
            const int   j     = kb + s * 16 + Lm;
            const float basef = (float)(i0 - j);
            float p = __builtin_amdgcn_exp2f(LOG2GAMMA * basef);    // gamma^(i0-j)
            float q = __builtin_amdgcn_exp2f(-LOG2GAMMA * basef);   // gamma^(j-i0)
#pragma unroll
            for (int v = 0; v < 8; ++v) {
                const int   M = v + 8 * Lh;
                const float d = fminf(p, q);                        // gamma^|i-j|
                sS[wave][M * SSTR + s * 16 + Lm] = (_Float16)(sc[v] * d);
                p *= GAMMA_F;
                q *= INV_GAMMA_F;
            }
        }
        __syncthreads();  // sS visible to all lanes

        // --- O += S(16x32) @ V(32x64): A from sS, four B tiles from sV ---
        v16h as;
#pragma unroll
        for (int v = 0; v < 8; ++v) {
#pragma unroll
            for (int h = 0; h < 2; ++h) {
                const int kk = 16 * Lh + 2 * v + h;
                as[2 * v + h] = sS[wave][Lm * SSTR + kk];
            }
        }
#pragma unroll
        for (int n = 0; n < 4; ++n) {
            v16h bv;
#pragma unroll
            for (int v = 0; v < 8; ++v) {
#pragma unroll
                for (int h = 0; h < 2; ++h) {
                    const int kd = 16 * Lh + 2 * v + h;      // local key index
                    bv[2 * v + h] = sV[kd * KSTR + n * 16 + Lm];
                }
            }
            o[n] = __builtin_amdgcn_wmma_f32_16x16x32_f16(false, as, false, bv,
                                                          (short)0, o[n], false, false);
        }
    }

    // --- write O tile: C layout M = v + 8*(lane/16), N = lane%16 ---
    float* Ob = Out + ((size_t)b * SEQ + qbase + wave * 16) * HEAD;
#pragma unroll
    for (int n = 0; n < 4; ++n) {
#pragma unroll
        for (int v = 0; v < 8; ++v) {
            const int M = v + 8 * Lh;
            Ob[(size_t)M * HEAD + n * 16 + Lm] = o[n][v];
        }
    }
}

// ---------------------------------------------------------------------------
extern "C" void kernel_launch(void* const* d_in, const int* in_sizes, int n_in,
                              void* d_out, int out_size, void* d_ws, size_t ws_size,
                              hipStream_t stream)
{
    (void)in_sizes; (void)n_in; (void)out_size; (void)ws_size;

    const float* X  = (const float*)d_in[0];
    const float* WQ = (const float*)d_in[1];
    const float* WK = (const float*)d_in[2];
    const float* WV = (const float*)d_in[3];
    float* Out = (float*)d_out;

    const size_t nQKV = (size_t)BATCH * SEQ * HEAD;  // 1,048,576 elems
    _Float16* Qh = (_Float16*)d_ws;
    _Float16* Kh = Qh + nQKV;
    _Float16* Vh = Kh + nQKV;                        // total 6 MB of ws

    // Projections (f32 WMMA) + xPos -> f16 Q/K/V
    dim3 g1(BATCH * (SEQ / 16));
    retnet_proj_xpos<<<g1, 384, 0, stream>>>(X, WQ, WK, WV, Qh, Kh, Vh);

    // Fused retention attention
    dim3 g2(BATCH * (SEQ / 64));
    retnet_attn_wmma<<<g2, 128, 0, stream>>>(Qh, Kh, Vh, Out);
}
